// FP_Layer_PN2_5583457485363
// MI455X (gfx1250) — matrix operand order repaired and checked by reading, not compiled, not run on hardware
//
#include <hip/hip_runtime.h>
#include <hip/hip_bf16.h>
#include <stdint.h>

#define B_    16
#define NL_   4096
#define NH_   1024
#define CL_   128
#define CH_   256
#define C0_   384      // CL+CH = GEMM0 K
#define OC_   256      // output channels (both layers)
#define BN_EPS_ 1e-5f
#define NTOT_ 65536    // B_*NL_ reduction count for BN stats

typedef __attribute__((ext_vector_type(16))) __bf16 v16bf;
typedef __attribute__((ext_vector_type(8)))  float  v8f;

union Frag32 { uint4 q[2]; v16bf v; };

__device__ __forceinline__ unsigned short f2bf(float f) {
  unsigned int u = __float_as_uint(f);
  u += 0x7FFFu + ((u >> 16) & 1u);          // round-to-nearest-even
  return (unsigned short)(u >> 16);
}

// ---- CDNA5 async global->LDS copy (GLOBAL_LOAD_ASYNC_TO_LDS_B128) -----
// VGLOBAL encoding: VDST = per-lane LDS byte address, VADDR = 64-bit global
// address. Tracked by ASYNCcnt. Low 32 bits of a flat LDS pointer are the
// wave-relative LDS offset (flat->LDS aperture truncation, ISA 10.2).
__device__ __forceinline__ void async_copy_b128(const void* gptr, void* lptr) {
  unsigned long long ga = (unsigned long long)(uintptr_t)gptr;
  unsigned int la = (unsigned int)(uintptr_t)lptr;
  asm volatile("global_load_async_to_lds_b128 %0, %1, off"
               :: "v"(la), "v"(ga) : "memory");
}
__device__ __forceinline__ void wait_async0() {
#if __has_builtin(__builtin_amdgcn_s_wait_asynccnt)
  __builtin_amdgcn_s_wait_asynccnt(0);
#else
  asm volatile("s_wait_asynccnt 0x0" ::: "memory");
#endif
}

// ---- WMMA fragment loaders from LDS -----------------------------------
// A tile in LDS: As[m][k], 64 rows x 32 k (bf16). 16-bit A 16x32 layout:
// lane L (m=L&15, h=L>>4): elems 0..7 = K[8h..8h+7], elems 8..15 = K[16+8h..]
__device__ __forceinline__ v16bf load_a_frag(const unsigned short (*As)[32], int mbase) {
  const int lane = threadIdx.x & 31;
  const int m = lane & 15, h = lane >> 4;
  const uint4* p = reinterpret_cast<const uint4*>(&As[mbase + m][0]);
  Frag32 f; f.q[0] = p[h]; f.q[1] = p[2 + h];
  return f.v;
}
// B tile in LDS TRANSPOSED: Bs[n][k], 128 rows x 32 k (bf16). 16-bit B 32x16:
// lane L (c=L&15, h=L>>4): elems 0..7 = Bs[c][8h..8h+7], elems 8..15 = Bs[c][16+8h..]
__device__ __forceinline__ v16bf load_b_frag(const unsigned short (*Bs)[32], int nbase) {
  const int lane = threadIdx.x & 31;
  const int c = lane & 15, h = lane >> 4;
  const uint4* p = reinterpret_cast<const uint4*>(&Bs[nbase + c][0]);
  Frag32 f; f.q[0] = p[h]; f.q[1] = p[2 + h];
  return f.v;
}

// ---- Kernel 1: 3-NN + inverse-distance interp + concat -> xcat (bf16) --
__global__ __launch_bounds__(256) void knn_interp_concat(
    const float* __restrict__ xyz_low, const float* __restrict__ xyz_high,
    const float* __restrict__ feat_low, const float* __restrict__ feat_high,
    unsigned short* __restrict__ xcat) {
  __shared__ float shx[NH_], shy[NH_], shz[NH_];
  const int tid = threadIdx.x;
  const int b = blockIdx.y;
  const int n = blockIdx.x * 256 + tid;
  const float* xh = xyz_high + (size_t)b * NH_ * 3;
  for (int i = tid; i < NH_ * 3; i += 256) {
    float v = xh[i];
    int j = i / 3, c = i - 3 * j;
    if (c == 0) shx[j] = v; else if (c == 1) shy[j] = v; else shz[j] = v;
  }
  __syncthreads();
  const float* xl = xyz_low + ((size_t)b * NL_ + n) * 3;
  const float px = xl[0], py = xl[1], pz = xl[2];
  float d0 = 3.4e38f, d1 = 3.4e38f, d2 = 3.4e38f;
  int i0 = 0, i1 = 0, i2 = 0;
  for (int j = 0; j < NH_; ++j) {
    float dx = px - shx[j], dy = py - shy[j], dz = pz - shz[j];
    float d = dx * dx + dy * dy + dz * dz;
    if (d < d2) {
      if (d < d1) {
        d2 = d1; i2 = i1;
        if (d < d0) { d1 = d0; i1 = i0; d0 = d; i0 = j; }
        else        { d1 = d;  i1 = j; }
      } else { d2 = d; i2 = j; }
    }
  }
  float w0 = 1.0f / fmaxf(sqrtf(fmaxf(d0, 0.f)), 1e-8f);
  float w1 = 1.0f / fmaxf(sqrtf(fmaxf(d1, 0.f)), 1e-8f);
  float w2 = 1.0f / fmaxf(sqrtf(fmaxf(d2, 0.f)), 1e-8f);
  const float wsum = 1.0f / (w0 + w1 + w2);
  w0 *= wsum; w1 *= wsum; w2 *= wsum;
  const float* FH = feat_high + (size_t)b * CH_ * NH_;
  unsigned short* xo = xcat + (size_t)b * C0_ * NL_ + n;
  for (int c = 0; c < CH_; ++c) {
    const float* row = FH + (size_t)c * NH_;
    float f = w0 * row[i0] + w1 * row[i1] + w2 * row[i2];
    xo[(size_t)c * NL_] = f2bf(f);
  }
  const float* FL = feat_low + (size_t)b * CL_ * NL_ + n;
  for (int c = 0; c < CL_; ++c)
    xo[(size_t)(CH_ + c) * NL_] = f2bf(FL[(size_t)c * NL_]);
}

// ---- Kernel 2: convert conv weights to bf16 ---------------------------
__global__ __launch_bounds__(256) void convert_weights(
    const float* __restrict__ W0, const float* __restrict__ W1,
    unsigned short* __restrict__ W0b, unsigned short* __restrict__ W1b) {
  int i = blockIdx.x * 256 + threadIdx.x;
  if (i < OC_ * C0_) W0b[i] = f2bf(W0[i]);
  int j = i - OC_ * C0_;
  if (j >= 0 && j < OC_ * OC_) W1b[j] = f2bf(W1[j]);
}

// ---- Kernel 3: GEMM0 (W0[256x384] x xcat[384x4096]) + bias + stats ----
__global__ __launch_bounds__(256) void gemm0_bias_stats(
    const unsigned short* __restrict__ Wb,    // [256][384] bf16
    const unsigned short* __restrict__ X,     // [B][384][4096] bf16
    const float* __restrict__ bias,
    float* __restrict__ Y,                    // [B][256][4096] f32
    float* __restrict__ gsum, float* __restrict__ gsq) {
  __shared__ __align__(16) unsigned short As[64][32];
  __shared__ __align__(16) unsigned short Bs[128][32];
  __shared__ float biasS[64], redS[64], redQ[64];
  const int tid = threadIdx.x;
  const int bb = blockIdx.z;
  const int M0 = blockIdx.y * 64;
  const int N0 = blockIdx.x * 128;
  if (tid < 64) { biasS[tid] = bias[M0 + tid]; redS[tid] = 0.f; redQ[tid] = 0.f; }

  const int wave = tid >> 5;
  const int wm = (wave & 1) * 32;
  const int wn = (wave >> 1) * 32;
  v8f acc[2][2] = {};

  const int am = tid >> 2;            // A row 0..63
  const int ak = (tid & 3) * 8;       // A k   0,8,16,24
  const int bk = (tid >> 4) << 1;     // B k   0,2,..,30  (2 rows per thread)
  const int bp = (tid & 15) * 8;      // B col 0,8,..,120 (8 cols per thread)

  const unsigned short* Ablk = Wb + (size_t)M0 * C0_;
  const unsigned short* Bblk = X + (size_t)bb * C0_ * NL_ + N0;

  for (int kk = 0; kk < C0_; kk += 32) {
    // A tile: async DMA global->LDS, 16B per lane, no VGPR round trip
    async_copy_b128(Ablk + (size_t)am * C0_ + kk + ak, &As[am][ak]);

    // B tile: 2(k) x 8(n) per thread, pack k-pairs -> dword LDS stores
    const unsigned short* r0 = Bblk + (size_t)(kk + bk) * NL_ + bp;
    const unsigned short* r1 = r0 + NL_;
    if (kk + 32 < C0_) __builtin_prefetch(r0 + 32 * NL_, 0, 1);
    unsigned short t0[8], t1[8];
    *reinterpret_cast<uint4*>(t0) = *reinterpret_cast<const uint4*>(r0);
    *reinterpret_cast<uint4*>(t1) = *reinterpret_cast<const uint4*>(r1);
#pragma unroll
    for (int i = 0; i < 8; ++i) {
      unsigned int w = (unsigned int)t0[i] | ((unsigned int)t1[i] << 16);
      *reinterpret_cast<unsigned int*>(&Bs[bp + i][bk]) = w;
    }
    wait_async0();
    __syncthreads();

    v16bf a0 = load_a_frag(As, wm);
    v16bf a1 = load_a_frag(As, wm + 16);
    v16bf bf0 = load_b_frag(Bs, wn);
    v16bf bf1 = load_b_frag(Bs, wn + 16);
    acc[0][0] = __builtin_amdgcn_wmma_f32_16x16x32_bf16(false, a0, false, bf0, (short)0, acc[0][0], false, false);
    acc[0][1] = __builtin_amdgcn_wmma_f32_16x16x32_bf16(false, a0, false, bf1, (short)0, acc[0][1], false, false);
    acc[1][0] = __builtin_amdgcn_wmma_f32_16x16x32_bf16(false, a1, false, bf0, (short)0, acc[1][0], false, false);
    acc[1][1] = __builtin_amdgcn_wmma_f32_16x16x32_bf16(false, a1, false, bf1, (short)0, acc[1][1], false, false);
    __syncthreads();
  }

  const int lane = tid & 31;
  const int h = lane >> 4, cL = lane & 15;
  float* Yb = Y + ((size_t)bb * OC_ + M0) * NL_ + N0;
#pragma unroll
  for (int i = 0; i < 2; ++i)
#pragma unroll
    for (int j = 0; j < 2; ++j) {
      const int ncol = wn + 16 * j + cL;
#pragma unroll
      for (int r = 0; r < 8; ++r) {
        const int row = wm + 16 * i + r + 8 * h;   // C/D layout: VGPR r -> row r+8h
        float v = acc[i][j][r] + biasS[row];
        Yb[(size_t)row * NL_ + ncol] = v;
        atomicAdd(&redS[row], v);
        atomicAdd(&redQ[row], v * v);
      }
    }
  __syncthreads();
  if (tid < 64) {
    atomicAdd(&gsum[M0 + tid], redS[tid]);
    atomicAdd(&gsq[M0 + tid], redQ[tid]);
  }
}

// ---- Kernel 4: GEMM1 with BN0+ReLU fused into B-tile load + stats -----
__global__ __launch_bounds__(256) void gemm1_bn_stats(
    const unsigned short* __restrict__ Wb,    // [256][256] bf16
    const float* __restrict__ Y0,             // [B][256][4096] pre-BN layer0
    const float* __restrict__ bias,
    const float* __restrict__ gamma0, const float* __restrict__ beta0,
    const float* __restrict__ sum0, const float* __restrict__ sq0,
    float* __restrict__ Y1,                   // d_out used as pre-BN scratch
    float* __restrict__ gsum, float* __restrict__ gsq) {
  __shared__ __align__(16) unsigned short As[64][32];
  __shared__ __align__(16) unsigned short Bs[128][32];
  __shared__ float biasS[64], redS[64], redQ[64];
  __shared__ float bnA[256], bnC[256];
  const int tid = threadIdx.x;
  {
    float s = sum0[tid], q = sq0[tid];
    float mean = s * (1.0f / NTOT_);
    float var = q * (1.0f / NTOT_) - mean * mean;
    float a = gamma0[tid] * rsqrtf(var + BN_EPS_);
    bnA[tid] = a;
    bnC[tid] = beta0[tid] - mean * a;
  }
  const int bb = blockIdx.z;
  const int M0 = blockIdx.y * 64;
  const int N0 = blockIdx.x * 128;
  if (tid < 64) { biasS[tid] = bias[M0 + tid]; redS[tid] = 0.f; redQ[tid] = 0.f; }
  __syncthreads();

  const int wave = tid >> 5;
  const int wm = (wave & 1) * 32;
  const int wn = (wave >> 1) * 32;
  v8f acc[2][2] = {};

  const int am = tid >> 2;
  const int ak = (tid & 3) * 8;
  const int bk = (tid >> 4) << 1;     // k pair base
  const int bp = (tid & 15) * 8;      // 8 columns

  const unsigned short* Ablk = Wb + (size_t)M0 * OC_;
  const float* Bblk = Y0 + (size_t)bb * OC_ * NL_ + N0;

  for (int kk = 0; kk < OC_; kk += 32) {
    async_copy_b128(Ablk + (size_t)am * OC_ + kk + ak, &As[am][ak]);

    const int k0 = kk + bk, k1 = k0 + 1;
    const float a0v = bnA[k0], c0v = bnC[k0];
    const float a1v = bnA[k1], c1v = bnC[k1];
    const float* r0 = Bblk + (size_t)k0 * NL_ + bp;
    const float* r1 = r0 + NL_;
    if (kk + 32 < OC_) __builtin_prefetch(r0 + 32 * NL_, 0, 1);
    float e0[8], e1[8];
    *reinterpret_cast<float4*>(&e0[0]) = reinterpret_cast<const float4*>(r0)[0];
    *reinterpret_cast<float4*>(&e0[4]) = reinterpret_cast<const float4*>(r0)[1];
    *reinterpret_cast<float4*>(&e1[0]) = reinterpret_cast<const float4*>(r1)[0];
    *reinterpret_cast<float4*>(&e1[4]) = reinterpret_cast<const float4*>(r1)[1];
#pragma unroll
    for (int i = 0; i < 8; ++i) {
      unsigned int lo = f2bf(fmaxf(fmaf(a0v, e0[i], c0v), 0.f));
      unsigned int hi = f2bf(fmaxf(fmaf(a1v, e1[i], c1v), 0.f));
      *reinterpret_cast<unsigned int*>(&Bs[bp + i][bk]) = lo | (hi << 16);
    }
    wait_async0();
    __syncthreads();

    v16bf a0 = load_a_frag(As, wm);
    v16bf a1 = load_a_frag(As, wm + 16);
    v16bf bf0 = load_b_frag(Bs, wn);
    v16bf bf1 = load_b_frag(Bs, wn + 16);
    acc[0][0] = __builtin_amdgcn_wmma_f32_16x16x32_bf16(false, a0, false, bf0, (short)0, acc[0][0], false, false);
    acc[0][1] = __builtin_amdgcn_wmma_f32_16x16x32_bf16(false, a0, false, bf1, (short)0, acc[0][1], false, false);
    acc[1][0] = __builtin_amdgcn_wmma_f32_16x16x32_bf16(false, a1, false, bf0, (short)0, acc[1][0], false, false);
    acc[1][1] = __builtin_amdgcn_wmma_f32_16x16x32_bf16(false, a1, false, bf1, (short)0, acc[1][1], false, false);
    __syncthreads();
  }

  const int lane = tid & 31;
  const int h = lane >> 4, cL = lane & 15;
  float* Yb = Y1 + ((size_t)bb * OC_ + M0) * NL_ + N0;
#pragma unroll
  for (int i = 0; i < 2; ++i)
#pragma unroll
    for (int j = 0; j < 2; ++j) {
      const int ncol = wn + 16 * j + cL;
#pragma unroll
      for (int r = 0; r < 8; ++r) {
        const int row = wm + 16 * i + r + 8 * h;
        float v = acc[i][j][r] + biasS[row];
        Yb[(size_t)row * NL_ + ncol] = v;
        atomicAdd(&redS[row], v);
        atomicAdd(&redQ[row], v * v);
      }
    }
  __syncthreads();
  if (tid < 64) {
    atomicAdd(&gsum[M0 + tid], redS[tid]);
    atomicAdd(&gsq[M0 + tid], redQ[tid]);
  }
}

// ---- Kernel 5: final BN1 + ReLU in place on d_out ---------------------
__global__ __launch_bounds__(256) void bn_relu_final(
    float* __restrict__ Y, const float* __restrict__ gamma,
    const float* __restrict__ beta,
    const float* __restrict__ gsum, const float* __restrict__ gsq) {
  const size_t i4 = (size_t)blockIdx.x * 256 + threadIdx.x;
  const size_t base = i4 * 4;
  const int ch = (int)((base >> 12) & (OC_ - 1));   // 4096 cols per channel
  float s = gsum[ch], q = gsq[ch];
  float mean = s * (1.0f / NTOT_);
  float var = q * (1.0f / NTOT_) - mean * mean;
  float a = gamma[ch] * rsqrtf(var + BN_EPS_);
  float c = beta[ch] - mean * a;
  float4* p = reinterpret_cast<float4*>(Y) + i4;
  float4 v = *p;
  v.x = fmaxf(fmaf(a, v.x, c), 0.f);
  v.y = fmaxf(fmaf(a, v.y, c), 0.f);
  v.z = fmaxf(fmaf(a, v.z, c), 0.f);
  v.w = fmaxf(fmaf(a, v.w, c), 0.f);
  *p = v;
}

// ---- Workspace layout --------------------------------------------------
// [0,        196608)  W0 bf16  (256*384*2)
// [196608,   327680)  W1 bf16  (256*256*2)
// [327680,   331776)  stats: sum0|sq0|sum1|sq1 (4*256 f32)
// [331776, 50663424)  xcat bf16 (16*384*4096*2)
// [50663424,117772288) Y0 f32  (16*256*4096*4)

extern "C" void kernel_launch(void* const* d_in, const int* in_sizes, int n_in,
                              void* d_out, int out_size, void* d_ws, size_t ws_size,
                              hipStream_t stream) {
  (void)in_sizes; (void)n_in; (void)out_size; (void)ws_size;
  const float* xyz_low   = (const float*)d_in[0];
  const float* xyz_high  = (const float*)d_in[1];
  const float* feat_low  = (const float*)d_in[2];
  const float* feat_high = (const float*)d_in[3];
  const float* W0  = (const float*)d_in[4];
  const float* b0  = (const float*)d_in[5];
  const float* g0  = (const float*)d_in[6];
  const float* be0 = (const float*)d_in[7];
  const float* W1  = (const float*)d_in[8];
  const float* b1  = (const float*)d_in[9];
  const float* g1  = (const float*)d_in[10];
  const float* be1 = (const float*)d_in[11];

  char* ws = (char*)d_ws;
  unsigned short* W0b  = (unsigned short*)(ws + 0);
  unsigned short* W1b  = (unsigned short*)(ws + 196608);
  float* stats         = (float*)(ws + 327680);
  float* sum0 = stats, * sq0 = stats + 256, * sum1 = stats + 512, * sq1 = stats + 768;
  unsigned short* xcat = (unsigned short*)(ws + 331776);
  float* Y0            = (float*)(ws + 50663424);
  float* out           = (float*)d_out;

  hipMemsetAsync(stats, 0, 4 * 256 * sizeof(float), stream);
  convert_weights<<<640, 256, 0, stream>>>(W0, W1, W0b, W1b);
  dim3 gknn(NL_ / 256, B_);
  knn_interp_concat<<<gknn, 256, 0, stream>>>(xyz_low, xyz_high, feat_low, feat_high, xcat);
  dim3 ggemm(NL_ / 128, OC_ / 64, B_);
  gemm0_bias_stats<<<ggemm, 256, 0, stream>>>(W0b, xcat, b0, Y0, sum0, sq0);
  gemm1_bn_stats<<<ggemm, 256, 0, stream>>>(W1b, Y0, b1, g0, be0, sum0, sq0, out, sum1, sq1);
  bn_relu_final<<<(B_ * OC_ * NL_) / 4 / 256, 256, 0, stream>>>(out, g1, be1, sum1, sq1);
}
